// GNN_23038204576079
// MI455X (gfx1250) — matrix operand order repaired and checked by reading, not compiled, not run on hardware
//
#include <hip/hip_runtime.h>
#include <hip/hip_bf16.h>

#define D_MODEL 96
#define N_TILES 6            // D_MODEL / 16
#define F_IN    16

typedef __attribute__((ext_vector_type(16))) __bf16 v16bf;
typedef __attribute__((ext_vector_type(8)))  float  v8f;

// ---------------------------------------------------------------------------
// Pack weight matrix W[K, 96] (f32) into per-lane bf16 B-fragments:
//   P[((kc*6 + t)*32 + lane)*16 + i]
// lane: half = lane>>4, m = lane&15; element i ->
//   kb = kc*32 + 8*half + i + (i>=8 ? 8 : 0) ; col = t*16 + m
// Zero-padded beyond K, so the GEMM loads are unconditional.
// ---------------------------------------------------------------------------
__global__ void pack_w(const float* __restrict__ W, __bf16* __restrict__ P,
                       int K, int total) {
    int idx = blockIdx.x * blockDim.x + threadIdx.x;
    if (idx >= total) return;
    int i    = idx & 15;
    int rest = idx >> 4;
    int l    = rest & 31;
    int rest2 = rest >> 5;
    int t    = rest2 % N_TILES;
    int kc   = rest2 / N_TILES;
    int j    = i >> 1, odd = i & 1;
    int half = l >> 4, m = l & 15;
    int kb   = kc * 32 + 8 * half + i + ((i >= 8) ? 8 : 0);
    int col  = t * 16 + m;
    float v  = (kb < K) ? W[(long)kb * D_MODEL + col] : 0.0f;
    P[idx] = (__bf16)v;
}

// ---------------------------------------------------------------------------
// WMMA GEMM: C[M,96] = A[M,K] @ W[K,96] (+bias+leaky_relu if BIASACT).
// One wave computes a 16x96 row-stripe. K compile-time -> fully unrolled,
// branch-free. All 6 B fragments of a k-chunk loaded in one clause so the
// WMMAs can drain outstanding loads instead of stalling per tile.
// GUARD=false when M % 16 == 0 -> epilogue is pure coalesced stores.
// ---------------------------------------------------------------------------
template <int K, bool BIASACT, bool GUARD>
__global__ __launch_bounds__(32)
void gcn_wmma_gemm(const float* __restrict__ A, const __bf16* __restrict__ PB,
                   const float* __restrict__ bias, float* __restrict__ C,
                   int M)
{
    constexpr int KC = (K + 31) / 32;
    const int tileM = blockIdx.x;
    const int lane  = threadIdx.x & 31;
    const int half  = lane >> 4;
    const int m16   = lane & 15;

    int rowL = tileM * 16 + m16;
    if (rowL > M - 1) rowL = M - 1;                  // clamp: branch-free loads
    const float* __restrict__ arow = A + (long)rowL * K;
    const v16bf* __restrict__ pb   = (const v16bf*)PB + lane;

    v8f acc[N_TILES];
#pragma unroll
    for (int t = 0; t < N_TILES; ++t) acc[t] = (v8f){};

#pragma unroll
    for (int kc = 0; kc < KC; ++kc) {
        // ---- all 6 B fragments of this chunk: one clause of b128 loads
        v16bf b[N_TILES];
#pragma unroll
        for (int t = 0; t < N_TILES; ++t)
            b[t] = pb[(kc * N_TILES + t) * 32];
        // ---- A fragment: two 16-wide K groups, 8 contiguous floats per lane
        v16bf a;
#pragma unroll
        for (int i = 0; i < 8; ++i) {
            const int grp = (i < 4) ? 0 : 16;                   // compile-time
            const bool ok = (kc * 32 + grp + 16) <= K;          // compile-time
            int ka = kc * 32 + grp + 2 * (i & 3) + 8 * half;
            float a0 = ok ? arow[ka]     : 0.0f;
            float a1 = ok ? arow[ka + 1] : 0.0f;
            a[2 * i]     = (__bf16)a0;
            a[2 * i + 1] = (__bf16)a1;
        }
        // ---- 6 column tiles share this A fragment
#pragma unroll
        for (int t = 0; t < N_TILES; ++t)
            acc[t] = __builtin_amdgcn_wmma_f32_16x16x32_bf16(
                         false, a, false, b[t], (short)0, acc[t], false, false);
    }

    // ---- epilogue
#pragma unroll
    for (int t = 0; t < N_TILES; ++t) {
        int col = t * 16 + m16;
        float bv = BIASACT ? bias[col] : 0.0f;
#pragma unroll
        for (int r = 0; r < 8; ++r) {
            int orow = tileM * 16 + (half ? (8 + r) : r);
            float v = acc[t][r] + bv;
            if (BIASACT) v = (v > 0.0f) ? v : 0.01f * v;
            if (!GUARD || orow < M) C[(long)orow * D_MODEL + col] = v;
        }
    }
}

template <int K, bool BIASACT>
static void launch_gemm(const float* A, const __bf16* PB, const float* bias,
                        float* C, int M, hipStream_t stream) {
    int mTiles = (M + 15) / 16;
    if ((M & 15) == 0)
        gcn_wmma_gemm<K, BIASACT, false><<<mTiles, 32, 0, stream>>>(A, PB, bias, C, M);
    else
        gcn_wmma_gemm<K, BIASACT, true><<<mTiles, 32, 0, stream>>>(A, PB, bias, C, M);
}

// ---------------------------------------------------------------------------
// Degree / normalization
// ---------------------------------------------------------------------------
__global__ void init_deg(float* deg, int n) {
    int i = blockIdx.x * blockDim.x + threadIdx.x;
    if (i < n) deg[i] = 1.0f;                       // self-loop
}
__global__ void edge_deg(const int* __restrict__ ei, float* deg, int E) {
    int e = blockIdx.x * blockDim.x + threadIdx.x;
    if (e < E) atomicAdd(&deg[ei[E + e]], 1.0f);    // dst = ei[1][e]
}
__global__ void rsqrt_inplace(float* d, int n) {
    int i = blockIdx.x * blockDim.x + threadIdx.x;
    if (i < n) d[i] = rsqrtf(d[i]);
}

// ---------------------------------------------------------------------------
// Aggregation: agg = sum over in-edges of h[src]*dis[src]*dis[dst]
// Self-loop folded into the init (norm = dis[n]^2) -> full overwrite
// ---------------------------------------------------------------------------
__global__ void agg_init(const float* __restrict__ h, const float* __restrict__ dis,
                         float* __restrict__ agg, long total) {
    long i = (long)blockIdx.x * blockDim.x + threadIdx.x;
    if (i >= total) return;
    int n = (int)(i / D_MODEL);
    float s = dis[n];
    agg[i] = h[i] * s * s;
}

__global__ void edge_scatter(const int* __restrict__ ei, const float* __restrict__ h,
                             const float* __restrict__ dis, float* __restrict__ agg, int E) {
    long tid = (long)blockIdx.x * blockDim.x + threadIdx.x;
    long e = tid / 24;                 // 24 chunks of 4 floats cover D=96
    if (e >= E) return;
    int c  = (int)(tid % 24);
    int d0 = c * 4;
    int s  = ei[e];
    int t  = ei[E + e];
    float norm = dis[s] * dis[t];
    const float4 hv = *(const float4*)&h[(long)s * D_MODEL + d0];
    float* dst = &agg[(long)t * D_MODEL + d0];
    atomicAdd(dst + 0, hv.x * norm);
    atomicAdd(dst + 1, hv.y * norm);
    atomicAdd(dst + 2, hv.z * norm);
    atomicAdd(dst + 3, hv.w * norm);
}

__global__ void bias_act(const float* __restrict__ agg, const float* __restrict__ bias,
                         float* __restrict__ feat, long total) {
    long i = (long)blockIdx.x * blockDim.x + threadIdx.x;
    if (i >= total) return;
    float v = agg[i] + bias[i % D_MODEL];
    feat[i] = (v > 0.0f) ? v : 0.01f * v;
}

// ---------------------------------------------------------------------------
// Pooling: segment max (monotone-uint atomicMax) + segment mean
// ---------------------------------------------------------------------------
__device__ __forceinline__ unsigned encf(float x) {
    unsigned u = __float_as_uint(x);
    return (u & 0x80000000u) ? ~u : (u | 0x80000000u);
}
__device__ __forceinline__ float decf(unsigned u) {
    unsigned v = (u & 0x80000000u) ? (u & 0x7FFFFFFFu) : ~u;
    return __uint_as_float(v);
}

__global__ void pool_init(unsigned* gmax, float* gsum, float* cnt, int G) {
    int i = blockIdx.x * blockDim.x + threadIdx.x;
    int gd = G * D_MODEL;
    if (i < gd) { gmax[i] = encf(-INFINITY); gsum[i] = 0.0f; }
    if (i < G) cnt[i] = 0.0f;
}

__global__ void pool_scatter(const float* __restrict__ feat, const int* __restrict__ batch,
                             unsigned* gmax, float* gsum, float* cnt, long total) {
    long i = (long)blockIdx.x * blockDim.x + threadIdx.x;
    if (i >= total) return;
    int n = (int)(i / D_MODEL);
    int d = (int)(i % D_MODEL);
    int g = batch[n];
    float v = feat[i];
    atomicMax(&gmax[g * D_MODEL + d], encf(v));
    atomicAdd(&gsum[g * D_MODEL + d], v);
    if (d == 0) atomicAdd(&cnt[g], 1.0f);
}

__global__ void pool_final(const unsigned* __restrict__ gmax, const float* __restrict__ gsum,
                           const float* __restrict__ cnt, float* __restrict__ pooled, int G) {
    int i = blockIdx.x * blockDim.x + threadIdx.x;
    if (i >= G * D_MODEL) return;
    int g = i / D_MODEL, d = i % D_MODEL;
    float c = cnt[g]; c = c > 1.0f ? c : 1.0f;
    pooled[(long)g * 2 * D_MODEL + d]           = decf(gmax[i]);
    pooled[(long)g * 2 * D_MODEL + D_MODEL + d] = gsum[i] / c;
}

// Final [G,96] @ [96,1] + b
__global__ void mlp2(const float* __restrict__ hid, const float* __restrict__ W,
                     const float* __restrict__ b, float* __restrict__ out, int G) {
    int g = blockIdx.x * blockDim.x + threadIdx.x;
    if (g >= G) return;
    float s = b[0];
#pragma unroll 8
    for (int d = 0; d < D_MODEL; ++d) s += hid[(long)g * D_MODEL + d] * W[d];
    out[g] = s;
}

// ---------------------------------------------------------------------------
extern "C" void kernel_launch(void* const* d_in, const int* in_sizes, int n_in,
                              void* d_out, int out_size, void* d_ws, size_t ws_size,
                              hipStream_t stream) {
    const float* x     = (const float*)d_in[0];
    const int*   ei    = (const int*)d_in[1];
    const int*   batch = (const int*)d_in[2];
    const float* Wl[5] = { (const float*)d_in[3], (const float*)d_in[5],
                           (const float*)d_in[7], (const float*)d_in[9],
                           (const float*)d_in[11] };
    const float* bl[5] = { (const float*)d_in[4], (const float*)d_in[6],
                           (const float*)d_in[8], (const float*)d_in[10],
                           (const float*)d_in[12] };
    const float* Wo1 = (const float*)d_in[13];
    const float* bo1 = (const float*)d_in[14];
    const float* Wo2 = (const float*)d_in[15];
    const float* bo2 = (const float*)d_in[16];

    const int N = in_sizes[0] / F_IN;
    const int E = in_sizes[1] / 2;
    const int G = out_size;
    const long ND = (long)N * D_MODEL;

    // workspace partition (256B aligned)
    char* ws = (char*)d_ws;
    size_t off = 0;
    auto alloc = [&](size_t bytes) -> void* {
        void* p = ws + off;
        off = (off + bytes + 255) & ~(size_t)255;
        return p;
    };
    float*    bufH   = (float*)alloc(ND * sizeof(float));
    float*    bufAgg = (float*)alloc(ND * sizeof(float));
    float*    bufF   = (float*)alloc(ND * sizeof(float));
    float*    dis    = (float*)alloc((size_t)N * sizeof(float));
    __bf16*   packB  = (__bf16*)alloc((size_t)6 * N_TILES * 512 * sizeof(__bf16)); // KC<=6
    unsigned* gmax   = (unsigned*)alloc((size_t)G * D_MODEL * sizeof(unsigned));
    float*    gsum   = (float*)alloc((size_t)G * D_MODEL * sizeof(float));
    float*    cnt    = (float*)alloc((size_t)G * sizeof(float));
    float*    pooled = (float*)alloc((size_t)G * 2 * D_MODEL * sizeof(float));
    float*    hid    = (float*)alloc((size_t)G * D_MODEL * sizeof(float));
    (void)ws_size; (void)n_in;

    const int TB = 256;

    // degree -> dis = rsqrt(deg)
    init_deg<<<(N + TB - 1) / TB, TB, 0, stream>>>(dis, N);
    edge_deg<<<(E + TB - 1) / TB, TB, 0, stream>>>(ei, dis, E);
    rsqrt_inplace<<<(N + TB - 1) / TB, TB, 0, stream>>>(dis, N);

    // 5 GCN layers
    for (int l = 0; l < 5; ++l) {
        const float* A = (l == 0) ? x : bufF;
        const int    K = (l == 0) ? F_IN : D_MODEL;
        const int    KC = (K + 31) / 32;
        const int    ptotal = KC * N_TILES * 512;
        pack_w<<<(ptotal + TB - 1) / TB, TB, 0, stream>>>(Wl[l], packB, K, ptotal);
        if (l == 0)
            launch_gemm<F_IN, false>(A, packB, nullptr, bufH, N, stream);
        else
            launch_gemm<D_MODEL, false>(A, packB, nullptr, bufH, N, stream);
        agg_init<<<(int)((ND + TB - 1) / TB), TB, 0, stream>>>(bufH, dis, bufAgg, ND);
        long sthreads = (long)E * 24;
        edge_scatter<<<(int)((sthreads + TB - 1) / TB), TB, 0, stream>>>(
            ei, bufH, dis, bufAgg, E);
        bias_act<<<(int)((ND + TB - 1) / TB), TB, 0, stream>>>(bufAgg, bl[l], bufF, ND);
    }

    // pooling
    int gd = G * D_MODEL;
    pool_init<<<(gd + TB - 1) / TB, TB, 0, stream>>>(gmax, gsum, cnt, G);
    pool_scatter<<<(int)((ND + TB - 1) / TB), TB, 0, stream>>>(bufF, batch, gmax, gsum, cnt, ND);
    pool_final<<<(gd + TB - 1) / TB, TB, 0, stream>>>(gmax, gsum, cnt, pooled, G);

    // MLP head: [G,192]@[192,96] + bias + leaky_relu (WMMA), then [G,96]@[96,1]+b
    {
        const int K = 2 * D_MODEL, KC = K / 32;
        const int ptotal = KC * N_TILES * 512;
        pack_w<<<(ptotal + TB - 1) / TB, TB, 0, stream>>>(Wo1, packB, K, ptotal);
        launch_gemm<2 * D_MODEL, true>(pooled, packB, bo1, hid, G, stream);
    }
    mlp2<<<(G + 63) / 64, 64, 0, stream>>>(hid, Wo2, bo2, (float*)d_out, G);
}
